// DecoderBlock_46505905881222
// MI455X (gfx1250) — matrix operand order repaired and checked by reading, not compiled, not run on hardware
//
#include <hip/hip_runtime.h>
#include <stdint.h>

#define E_DIM   1024
#define HID_DIM 4096
#define NHEAD   16
#define HDIM    64
#define TSEQ    1024
#define BATCH   4
#define BTOK    (BATCH * TSEQ)

typedef __attribute__((ext_vector_type(16))) __bf16 v16bf;
typedef __attribute__((ext_vector_type(8)))  float  v8f;
typedef __attribute__((ext_vector_type(4)))  unsigned int v4u;
typedef __attribute__((ext_vector_type(8)))  int v8i;
typedef __attribute__((ext_vector_type(4)))  int v4i;

union FragAB {
    v16bf v;
    uint4 q[2];
    unsigned short s[16];
};

static __device__ __forceinline__ unsigned short f2bfbits(float f) {
    union { float f; uint32_t u; } a;
    a.f = f;
    uint32_t r = a.u + 0x7fffu + ((a.u >> 16) & 1u);  // round-to-nearest-even
    return (unsigned short)(r >> 16);
}

// ---------------------------------------------------------------------------
// Tensor Data Mover: 2D tile (bf16 elements) global -> LDS, contiguous in LDS.
// Descriptor per CDNA5 ISA section 8 (D# groups). One issue per workgroup
// (TDM ignores EXEC; descriptor values are block-uniform -> SGPRs).
// tile_w/tile_h in elements; row_stride in elements.
// Toolchain uses the 6-arg builtin: (v4u g0, v8i g1, v4i g2, v4i g3, v8i, cpol)
// ---------------------------------------------------------------------------
static __device__ __forceinline__ void tdm_load_2d(unsigned lds_addr,
                                                   const void* gptr,
                                                   unsigned tile_w, unsigned tile_h,
                                                   unsigned long long row_stride) {
    const unsigned long long ga = (unsigned long long)gptr;
    v4u g0;
    g0.x = 1u;                                   // count = 1 (valid descriptor)
    g0.y = lds_addr;                             // LDS byte address
    g0.z = (unsigned)(ga & 0xffffffffu);         // global_addr[31:0]
    g0.w = (unsigned)((ga >> 32) & 0x01ffffffu)  // global_addr[56:32]
           | (2u << 30);                         // type = 2 ("image")
    const unsigned td0 = 0x40000000u;            // huge tensor dims: never OOB-clamp
    const unsigned td1 = 0x40000000u;
    v8i g1;
    g1[0] = (int)(1u << 16);                                       // data_size = 1 (2 bytes)
    g1[1] = (int)((td0 & 0xffffu) << 16);                          // tensor_dim0[15:0]
    g1[2] = (int)(((td0 >> 16) & 0xffffu) | ((td1 & 0xffffu) << 16));
    g1[3] = (int)(((td1 >> 16) & 0xffffu) | ((tile_w & 0xffffu) << 16));  // tile_dim0
    g1[4] = (int)(tile_h & 0xffffu);                               // tile_dim1; tile_dim2=0
    g1[5] = (int)(unsigned)(row_stride & 0xffffffffu);             // tensor_dim0_stride
    g1[6] = (int)(unsigned)((row_stride >> 32) & 0xffffu);
    g1[7] = 0;
    v4i g2 = {0, 0, 0, 0};
    v4i g3 = {0, 0, 0, 0};
    v8i g4 = {0, 0, 0, 0, 0, 0, 0, 0};
    __builtin_amdgcn_tensor_load_to_lds(g0, g1, g2, g3, g4, 0);
}

// ---------------------------------------------------------------------------
// fp32 -> bf16 convert (vectorized, optional scale)
// ---------------------------------------------------------------------------
__global__ __launch_bounds__(256) void k_cvt_bf16(const float* __restrict__ src,
                                                  unsigned short* __restrict__ dst,
                                                  int n4, float scale) {
    int i = blockIdx.x * 256 + threadIdx.x;
    if (i >= n4) return;
    float4 v = reinterpret_cast<const float4*>(src)[i];
    uint2 w;
    w.x = (uint32_t)f2bfbits(v.x * scale) | ((uint32_t)f2bfbits(v.y * scale) << 16);
    w.y = (uint32_t)f2bfbits(v.z * scale) | ((uint32_t)f2bfbits(v.w * scale) << 16);
    reinterpret_cast<uint2*>(dst)[i] = w;
}

// ---------------------------------------------------------------------------
// bf16 GEMM: C(f32, MxN) = A(bf16, MxK row-major) @ B(bf16, KxN row-major)
// flags: bit0 = add bias[col], bit1 = relu
// Block tile 128x128, BK=32, 256 threads = 8 waves (2 x 4), wave tile 64x32.
// A tile staged global->LDS by the TDM; B tile staged transposed by VALU.
// ---------------------------------------------------------------------------
#define GBM 128
#define GBN 128
#define GBK 32

__global__ __launch_bounds__(256) void k_gemm_bf16(const unsigned short* __restrict__ A,
                                                   const unsigned short* __restrict__ Bm,
                                                   float* __restrict__ C,
                                                   const float* __restrict__ bias,
                                                   int M, int N, int K, int flags) {
    __shared__ alignas(32) unsigned short Al[GBM * GBK];   // [m][k]
    __shared__ alignas(32) unsigned short Bl[GBN * GBK];   // [n][k] (transposed)

    const int tid  = threadIdx.x;
    const int lane = tid & 31;
    const int wave = tid >> 5;
    const int r    = lane & 15;
    const int half = lane >> 4;
    const int wm   = wave >> 2;   // 0..1
    const int wn   = wave & 3;    // 0..3
    const int m0   = blockIdx.y * GBM;
    const int n0   = blockIdx.x * GBN;

    const unsigned al_base = (unsigned)(size_t)(&Al[0]);

    v8f acc[4][2] = {};

    const int b_krow = tid >> 3;          // 0..31
    const int b_nb   = (tid & 7) * 16;    // 0..112

    for (int k0 = 0; k0 < K; k0 += GBK) {
        // A tile: 128 rows x 32 cols, row stride K -> contiguous LDS, via TDM
        if (wave == 0)
            tdm_load_2d(al_base, A + (size_t)m0 * K + k0, GBK, GBM, (unsigned long long)K);

        // B tile transposed -> [n][k] (manual: needs transpose)
        const unsigned short* bg = Bm + (size_t)(k0 + b_krow) * N + n0 + b_nb;
        FragAB tb;
        tb.q[0] = *reinterpret_cast<const uint4*>(bg);
        tb.q[1] = *reinterpret_cast<const uint4*>(bg + 8);
#pragma unroll
        for (int i = 0; i < 16; ++i)
            Bl[(b_nb + i) * GBK + b_krow] = tb.s[i];

        if (wave == 0)
            __builtin_amdgcn_s_wait_tensorcnt(0);
        __syncthreads();

        FragAB af[4];
#pragma unroll
        for (int mi = 0; mi < 4; ++mi) {
            const int row = wm * 64 + mi * 16 + r;
            af[mi].v = *reinterpret_cast<const v16bf*>(&Al[row * GBK + half * 16]);
        }
        v16bf bf[2];
#pragma unroll
        for (int ni = 0; ni < 2; ++ni) {
            const int col = wn * 32 + ni * 16 + r;
            bf[ni] = *reinterpret_cast<const v16bf*>(&Bl[col * GBK + half * 16]);
        }
#pragma unroll
        for (int mi = 0; mi < 4; ++mi)
#pragma unroll
            for (int ni = 0; ni < 2; ++ni)
                acc[mi][ni] = __builtin_amdgcn_wmma_f32_16x16x32_bf16(
                    false, af[mi].v, false, bf[ni], (short)0, acc[mi][ni], false, false);
        __syncthreads();
    }

#pragma unroll
    for (int mi = 0; mi < 4; ++mi)
#pragma unroll
        for (int ni = 0; ni < 2; ++ni) {
            const int col = n0 + wn * 32 + ni * 16 + r;
            float bv = (flags & 1) ? bias[col] : 0.0f;
#pragma unroll
            for (int j = 0; j < 8; ++j) {
                const int row = m0 + wm * 64 + mi * 16 + half * 8 + j;
                float v = acc[mi][ni][j] + bv;
                if (flags & 2) v = fmaxf(v, 0.0f);
                C[(size_t)row * N + col] = v;
            }
        }
}

// ---------------------------------------------------------------------------
// Flash attention: one block per (q-block of 64 rows, head, batch).
// 128 threads = 4 waves; wave w owns q rows [w*16, w*16+16). Head dim 64.
// Q/K/V are bf16 [B*T][E] with head offset h*64; Q,K pre-scaled by E^-0.25.
// K tile staged by TDM; V tile staged transposed by VALU.
// ---------------------------------------------------------------------------
__global__ __launch_bounds__(128) void k_flash_attn(const unsigned short* __restrict__ Q,
                                                    const unsigned short* __restrict__ Km,
                                                    const unsigned short* __restrict__ V,
                                                    unsigned short* __restrict__ O,
                                                    int Tq, int Tc, int causal) {
    __shared__ alignas(32) unsigned short Kl[64 * 64];     // [kv][s]
    __shared__ alignas(32) unsigned short VT[64 * 64];     // [s][kv]
    __shared__ alignas(32) unsigned short Pl[4][16 * 64];  // per-wave P staging [q][kv]

    const int tid  = threadIdx.x;
    const int lane = tid & 31;
    const int wave = tid >> 5;
    const int r    = lane & 15;
    const int half = lane >> 4;
    const int qblk = blockIdx.x;
    const int h    = blockIdx.y;
    const int b    = blockIdx.z;

    const unsigned kl_base = (unsigned)(size_t)(&Kl[0]);

    // Q A-fragments (16 rows x 64 s), loaded once
    const int qrow0 = qblk * 64 + wave * 16;
    const unsigned short* qp = Q + ((size_t)(b * Tq + qrow0 + r) * E_DIM) + h * HDIM;
    FragAB qf[2];
#pragma unroll
    for (int ks = 0; ks < 2; ++ks) {
        qf[ks].v = *reinterpret_cast<const v16bf*>(qp + ks * 32 + half * 16);
    }

    float mrow[8], lrow[8];
    v8f oacc[4] = {};
#pragma unroll
    for (int j = 0; j < 8; ++j) { mrow[j] = -1e30f; lrow[j] = 0.0f; }

    const int nkv    = causal ? (qblk + 1) : (Tc / 64);
    const int ld_row = tid >> 1;          // 0..63
    const int ld_c0  = (tid & 1) * 32;    // 0 / 32

    for (int kb = 0; kb < nkv; ++kb) {
        const int kv0 = kb * 64;
        // K tile (64x64, row stride E) via TDM
        if (wave == 0)
            tdm_load_2d(kl_base, Km + ((size_t)(b * Tc + kv0) * E_DIM) + h * HDIM,
                        64, 64, (unsigned long long)E_DIM);
        // V tile transposed (manual)
        {
            const unsigned short* vg = V + ((size_t)(b * Tc + kv0 + ld_row) * E_DIM) + h * HDIM + ld_c0;
            FragAB t0, t1;
            t0.q[0] = *reinterpret_cast<const uint4*>(vg);
            t0.q[1] = *reinterpret_cast<const uint4*>(vg + 8);
            t1.q[0] = *reinterpret_cast<const uint4*>(vg + 16);
            t1.q[1] = *reinterpret_cast<const uint4*>(vg + 24);
#pragma unroll
            for (int i = 0; i < 16; ++i) {
                VT[(ld_c0 + i) * 64 + ld_row]      = t0.s[i];
                VT[(ld_c0 + 16 + i) * 64 + ld_row] = t1.s[i];
            }
        }
        if (wave == 0)
            __builtin_amdgcn_s_wait_tensorcnt(0);
        __syncthreads();

        // S = Q K^T (16 q rows x 64 kv cols), fp32 accumulate
        v8f sfr[4];
#pragma unroll
        for (int n = 0; n < 4; ++n) {
            v8f c = {};
#pragma unroll
            for (int ks = 0; ks < 2; ++ks) {
                v16bf kf = *reinterpret_cast<const v16bf*>(&Kl[(n * 16 + r) * 64 + ks * 32 + half * 16]);
                c = __builtin_amdgcn_wmma_f32_16x16x32_bf16(false, qf[ks].v, false, kf, (short)0, c, false, false);
            }
            sfr[n] = c;
        }

        if (causal && kb == qblk) {
#pragma unroll
            for (int n = 0; n < 4; ++n)
#pragma unroll
                for (int j = 0; j < 8; ++j) {
                    const int qi = qrow0 + half * 8 + j;
                    const int ki = kv0 + n * 16 + r;
                    if (ki > qi) sfr[n][j] = -1e30f;
                }
        }

        // online softmax (rows shared by the 16 lanes of each half-group)
        float mnew[8], pscale[8];
#pragma unroll
        for (int j = 0; j < 8; ++j) {
            float mx = fmaxf(fmaxf(sfr[0][j], sfr[1][j]), fmaxf(sfr[2][j], sfr[3][j]));
            mx = fmaxf(mx, __shfl_xor(mx, 1));
            mx = fmaxf(mx, __shfl_xor(mx, 2));
            mx = fmaxf(mx, __shfl_xor(mx, 4));
            mx = fmaxf(mx, __shfl_xor(mx, 8));
            mnew[j]   = fmaxf(mrow[j], mx);
            pscale[j] = __expf(mrow[j] - mnew[j]);
            mrow[j]   = mnew[j];
        }
        unsigned short* Pw = Pl[wave];
        float rsum[8];
#pragma unroll
        for (int j = 0; j < 8; ++j) rsum[j] = 0.0f;
#pragma unroll
        for (int n = 0; n < 4; ++n)
#pragma unroll
            for (int j = 0; j < 8; ++j) {
                float p = __expf(sfr[n][j] - mnew[j]);
                rsum[j] += p;
                Pw[(half * 8 + j) * 64 + n * 16 + r] = f2bfbits(p);
            }
#pragma unroll
        for (int j = 0; j < 8; ++j) {
            float s = rsum[j];
            s += __shfl_xor(s, 1);
            s += __shfl_xor(s, 2);
            s += __shfl_xor(s, 4);
            s += __shfl_xor(s, 8);
            lrow[j] = lrow[j] * pscale[j] + s;
        }
#pragma unroll
        for (int ns = 0; ns < 4; ++ns)
#pragma unroll
            for (int j = 0; j < 8; ++j)
                oacc[ns][j] *= pscale[j];

        // O += P @ V   (P relayout via per-wave LDS; DS ops are in-order per wave)
        FragAB pf[2];
#pragma unroll
        for (int ks = 0; ks < 2; ++ks) {
            pf[ks].v = *reinterpret_cast<const v16bf*>(&Pw[r * 64 + ks * 32 + half * 16]);
        }
#pragma unroll
        for (int ns = 0; ns < 4; ++ns) {
            v8f c = oacc[ns];
#pragma unroll
            for (int ks = 0; ks < 2; ++ks) {
                v16bf vf = *reinterpret_cast<const v16bf*>(&VT[(ns * 16 + r) * 64 + ks * 32 + half * 16]);
                c = __builtin_amdgcn_wmma_f32_16x16x32_bf16(false, pf[ks].v, false, vf, (short)0, c, false, false);
            }
            oacc[ns] = c;
        }
        __syncthreads();
    }

#pragma unroll
    for (int j = 0; j < 8; ++j) {
        const float inv = 1.0f / lrow[j];
        const int qi = qrow0 + half * 8 + j;
        unsigned short* op = O + ((size_t)(b * Tq + qi) * E_DIM) + h * HDIM;
#pragma unroll
        for (int ns = 0; ns < 4; ++ns)
            op[ns * 16 + r] = f2bfbits(oacc[ns][j] * inv);
    }
}

// ---------------------------------------------------------------------------
// Fused residual add + LayerNorm. One row (E=1024) per 256-thread block.
// Writes fp32 stream and optional bf16 copy.
// ---------------------------------------------------------------------------
__global__ __launch_bounds__(256) void k_resid_ln(const float* __restrict__ A,
                                                  const float* __restrict__ Bv,
                                                  const float* __restrict__ g,
                                                  const float* __restrict__ be,
                                                  float* __restrict__ outf,
                                                  unsigned short* __restrict__ outb) {
    __shared__ float s1[256];
    __shared__ float s2[256];
    const int row = blockIdx.x;
    const int tid = threadIdx.x;
    const float4 va = reinterpret_cast<const float4*>(A + (size_t)row * E_DIM)[tid];
    const float4 vb = reinterpret_cast<const float4*>(Bv + (size_t)row * E_DIM)[tid];
    const float x0 = va.x + vb.x, x1 = va.y + vb.y, x2 = va.z + vb.z, x3 = va.w + vb.w;
    s1[tid] = x0 + x1 + x2 + x3;
    s2[tid] = x0 * x0 + x1 * x1 + x2 * x2 + x3 * x3;
    __syncthreads();
    for (int st = 128; st > 0; st >>= 1) {
        if (tid < st) { s1[tid] += s1[tid + st]; s2[tid] += s2[tid + st]; }
        __syncthreads();
    }
    const float mean = s1[0] * (1.0f / E_DIM);
    const float var  = s2[0] * (1.0f / E_DIM) - mean * mean;
    const float rstd = rsqrtf(var + 1e-5f);
    const float4 vg = reinterpret_cast<const float4*>(g)[tid];
    const float4 vt = reinterpret_cast<const float4*>(be)[tid];
    float4 o;
    o.x = (x0 - mean) * rstd * vg.x + vt.x;
    o.y = (x1 - mean) * rstd * vg.y + vt.y;
    o.z = (x2 - mean) * rstd * vg.z + vt.z;
    o.w = (x3 - mean) * rstd * vg.w + vt.w;
    reinterpret_cast<float4*>(outf + (size_t)row * E_DIM)[tid] = o;
    if (outb) {
        uint2 w;
        w.x = (uint32_t)f2bfbits(o.x) | ((uint32_t)f2bfbits(o.y) << 16);
        w.y = (uint32_t)f2bfbits(o.z) | ((uint32_t)f2bfbits(o.w) << 16);
        reinterpret_cast<uint2*>(outb + (size_t)row * E_DIM)[tid] = w;
    }
}

// ---------------------------------------------------------------------------
extern "C" void kernel_launch(void* const* d_in, const int* in_sizes, int n_in,
                              void* d_out, int out_size, void* d_ws, size_t ws_size,
                              hipStream_t stream) {
    (void)in_sizes; (void)n_in; (void)out_size; (void)ws_size;

    const float* x     = (const float*)d_in[0];
    const float* ctx   = (const float*)d_in[1];
    const float* sa_wq = (const float*)d_in[2];
    const float* sa_wk = (const float*)d_in[3];
    const float* sa_wv = (const float*)d_in[4];
    const float* sa_wo = (const float*)d_in[5];
    const float* sa_bo = (const float*)d_in[6];
    const float* ca_wq = (const float*)d_in[7];
    const float* ca_wk = (const float*)d_in[8];
    const float* ca_wv = (const float*)d_in[9];
    const float* ca_wo = (const float*)d_in[10];
    const float* ca_bo = (const float*)d_in[11];
    const float* n1_g  = (const float*)d_in[12];
    const float* n1_b  = (const float*)d_in[13];
    const float* n2_g  = (const float*)d_in[14];
    const float* n2_b  = (const float*)d_in[15];
    const float* n3_g  = (const float*)d_in[16];
    const float* n3_b  = (const float*)d_in[17];
    const float* ff_w1 = (const float*)d_in[18];
    const float* ff_b1 = (const float*)d_in[19];
    const float* ff_w2 = (const float*)d_in[20];
    const float* ff_b2 = (const float*)d_in[21];

    char* ws = (char*)d_ws;
    size_t off = 0;
    auto take = [&](size_t bytes) -> char* {
        char* p = ws + off;
        off += (bytes + 255) & ~(size_t)255;
        return p;
    };

    float* t0 = (float*)take((size_t)BTOK * HID_DIM * 4);     // generic f32 GEMM output
    float* x1 = (float*)take((size_t)BTOK * E_DIM * 4);       // stream after LN1
    float* x2 = (float*)take((size_t)BTOK * E_DIM * 4);       // stream after LN2
    unsigned short* xb = (unsigned short*)take((size_t)BTOK * E_DIM * 2);
    unsigned short* cb = (unsigned short*)take((size_t)BTOK * E_DIM * 2);
    unsigned short* wb = (unsigned short*)take((size_t)E_DIM * HID_DIM * 2);
    unsigned short* qb = (unsigned short*)take((size_t)BTOK * E_DIM * 2);
    unsigned short* kb = (unsigned short*)take((size_t)BTOK * E_DIM * 2);
    unsigned short* vb = (unsigned short*)take((size_t)BTOK * E_DIM * 2);
    unsigned short* ob = (unsigned short*)take((size_t)BTOK * E_DIM * 2);
    unsigned short* hb = (unsigned short*)take((size_t)BTOK * HID_DIM * 2);

    const float qk_scale = 0.17677669529663687f;  // 1024^-0.25

    auto cvt = [&](const float* src, unsigned short* dst, size_t n, float scale) {
        int n4 = (int)(n / 4);
        k_cvt_bf16<<<dim3((n4 + 255) / 256), dim3(256), 0, stream>>>(src, dst, n4, scale);
    };
    auto gemm = [&](const unsigned short* A, const unsigned short* Bm, float* C,
                    const float* bias, int M, int N, int K, int flags) {
        k_gemm_bf16<<<dim3(N / GBN, M / GBM), dim3(256), 0, stream>>>(A, Bm, C, bias, M, N, K, flags);
    };

    const size_t nBE = (size_t)BTOK * E_DIM;
    const size_t nEE = (size_t)E_DIM * E_DIM;
    const size_t nEH = (size_t)E_DIM * HID_DIM;
    const size_t nBH = (size_t)BTOK * HID_DIM;

    // ---- self-attention ----
    cvt(x, xb, nBE, 1.0f);
    cvt(ctx, cb, nBE, 1.0f);

    cvt(sa_wq, wb, nEE, 1.0f);
    gemm(xb, wb, t0, nullptr, BTOK, E_DIM, E_DIM, 0);
    cvt(t0, qb, nBE, qk_scale);

    cvt(sa_wk, wb, nEE, 1.0f);
    gemm(xb, wb, t0, nullptr, BTOK, E_DIM, E_DIM, 0);
    cvt(t0, kb, nBE, qk_scale);

    cvt(sa_wv, wb, nEE, 1.0f);
    gemm(xb, wb, t0, nullptr, BTOK, E_DIM, E_DIM, 0);
    cvt(t0, vb, nBE, 1.0f);

    k_flash_attn<<<dim3(TSEQ / 64, NHEAD, BATCH), dim3(128), 0, stream>>>(
        qb, kb, vb, ob, TSEQ, TSEQ, 1);

    cvt(sa_wo, wb, nEE, 1.0f);
    gemm(ob, wb, t0, sa_bo, BTOK, E_DIM, E_DIM, 1);
    k_resid_ln<<<dim3(BTOK), dim3(256), 0, stream>>>(x, t0, n1_g, n1_b, x1, xb);

    // ---- cross-attention ----
    cvt(ca_wq, wb, nEE, 1.0f);
    gemm(xb, wb, t0, nullptr, BTOK, E_DIM, E_DIM, 0);
    cvt(t0, qb, nBE, qk_scale);

    cvt(ca_wk, wb, nEE, 1.0f);
    gemm(cb, wb, t0, nullptr, BTOK, E_DIM, E_DIM, 0);
    cvt(t0, kb, nBE, qk_scale);

    cvt(ca_wv, wb, nEE, 1.0f);
    gemm(cb, wb, t0, nullptr, BTOK, E_DIM, E_DIM, 0);
    cvt(t0, vb, nBE, 1.0f);

    k_flash_attn<<<dim3(TSEQ / 64, NHEAD, BATCH), dim3(128), 0, stream>>>(
        qb, kb, vb, ob, TSEQ, TSEQ, 0);

    cvt(ca_wo, wb, nEE, 1.0f);
    gemm(ob, wb, t0, ca_bo, BTOK, E_DIM, E_DIM, 1);
    k_resid_ln<<<dim3(BTOK), dim3(256), 0, stream>>>(x1, t0, n2_g, n2_b, x2, xb);

    // ---- FFN ----
    cvt(ff_w1, wb, nEH, 1.0f);
    gemm(xb, wb, t0, ff_b1, BTOK, HID_DIM, E_DIM, 3);  // bias + relu
    cvt(t0, hb, nBH, 1.0f);

    cvt(ff_w2, wb, nEH, 1.0f);
    gemm(hb, wb, t0, ff_b2, BTOK, E_DIM, HID_DIM, 1);
    k_resid_ln<<<dim3(BTOK), dim3(256), 0, stream>>>(x2, t0, n3_g, n3_b, (float*)d_out, nullptr);
}
